// VLAE_88957362635412
// MI455X (gfx1250) — matrix operand order repaired and checked by reading, not compiled, not run on hardware
//
#include <hip/hip_runtime.h>
#include <math.h>

typedef __attribute__((ext_vector_type(16))) _Float16 v16h;
typedef __attribute__((ext_vector_type(8)))  _Float16 v8h;
typedef __attribute__((ext_vector_type(8)))  float    v8f;
typedef __attribute__((ext_vector_type(4)))  float    v4f;

#define ZZ 32
#define HDIM 512
#define DDIM 784
#define KPAD 800      // 784 padded up to a multiple of 32
#define WSTRIDE 416   // 400-col W chunk padded to multiple of 32 (halves)
#define LOG2PI_C 1.8378770664093453f
#define ALPHA_C 0.5f

__device__ __forceinline__ v8f wmma16(v16h a, v16h b, v8f c) {
  return __builtin_amdgcn_wmma_f32_16x16x32_f16(false, a, false, b,
                                                (short)0, c, false, false);
}

// A fragment (16x32 f16) from row-major f32: lane reads two contiguous 8-float
// runs (K = 0..15 for lanes 0-15, 16..31 for lanes 16-31).
__device__ __forceinline__ v16h loadA_f32(const float* arow, int k0, int hi) {
  const v4f* p0 = (const v4f*)(arow + k0 + hi * 16);
  const v4f* p1 = (const v4f*)(arow + k0 + hi * 16 + 8);
  v4f f0 = p0[0], f1 = p0[1], f2 = p1[0], f3 = p1[1];
  v16h a;
#pragma unroll
  for (int j = 0; j < 4; ++j) {
    a[j]      = (_Float16)f0[j];
    a[4 + j]  = (_Float16)f1[j];
    a[8 + j]  = (_Float16)f2[j];
    a[12 + j] = (_Float16)f3[j];
  }
  return a;
}

// B fragment (32x16 f16) from row-major f32 "W[N,K]": 16 contiguous floats.
__device__ __forceinline__ v16h loadB_f32(const float* wrow, int k0, int hi) {
  const v4f* p = (const v4f*)(wrow + k0 + hi * 16);
  v4f f0 = p[0], f1 = p[1], f2 = p[2], f3 = p[3];
  v16h b;
#pragma unroll
  for (int j = 0; j < 4; ++j) {
    b[j]      = (_Float16)f0[j];
    b[4 + j]  = (_Float16)f1[j];
    b[8 + j]  = (_Float16)f2[j];
    b[12 + j] = (_Float16)f3[j];
  }
  return b;
}

// B fragment straight out of LDS f16 (16 contiguous halves, 32B-aligned).
__device__ __forceinline__ v16h loadB_h(const _Float16* brow, int k0, int hi) {
  return *(const v16h*)(brow + k0 + hi * 16);
}

// A fragment from LDS f16: two contiguous 8-half runs.
__device__ __forceinline__ v16h loadA_h(const _Float16* arow, int k0, int hi) {
  v8h lo = *(const v8h*)(arow + k0 + hi * 16);
  v8h hh = *(const v8h*)(arow + k0 + hi * 16 + 8);
  v16h a;
#pragma unroll
  for (int j = 0; j < 8; ++j) { a[j] = lo[j]; a[8 + j] = hh[j]; }
  return a;
}

// ---------------------------------------------------------------------------
// Pad-copy: dst[rows,Kp] = src[rows,K] zero-extended along K.
// ---------------------------------------------------------------------------
__global__ __launch_bounds__(256) void pad_copy(const float* __restrict__ src,
                                                float* __restrict__ dst,
                                                int K, int Kp) {
  int r = blockIdx.x;
  for (int k = threadIdx.x; k < Kp; k += 256)
    dst[(size_t)r * Kp + k] = (k < K) ? src[(size_t)r * K + k] : 0.f;
}

// ---------------------------------------------------------------------------
// WMMA GEMM: C[M,N] = act(A[M,K] @ W[N,K]^T + bias). One wave per 16x32 tile
// (one A fragment drives two v_wmma). Requires K % 32 == 0, M % 16, N % 16.
// ---------------------------------------------------------------------------
__global__ __launch_bounds__(32) void gemm_wmma(
    const float* __restrict__ A, const float* __restrict__ W,
    const float* __restrict__ bias, float* __restrict__ C,
    int M, int N, int K, int relu) {
  int ntn = (N + 31) >> 5;
  int tm = blockIdx.x / ntn, tn = blockIdx.x % ntn;
  int m0 = tm << 4, n0 = tn << 5;
  int lane = threadIdx.x & 31, lm = lane & 15, hi = lane >> 4;
  bool has2 = (n0 + 16) < N;
  const float* arow = A + (size_t)(m0 + lm) * K;
  const float* w0 = W + (size_t)(n0 + lm) * K;
  const float* w1 = W + (size_t)(n0 + (has2 ? 16 : 0) + lm) * K;
  v8f acc0 = {}, acc1 = {};
  for (int k0 = 0; k0 < K; k0 += 32) {
    __builtin_prefetch(arow + k0 + 64, 0, 0);
    v16h a = loadA_f32(arow, k0, hi);
    acc0 = wmma16(a, loadB_f32(w0, k0, hi), acc0);
    acc1 = wmma16(a, loadB_f32(w1, k0, hi), acc1);
  }
  float b0 = bias ? bias[n0 + lm] : 0.f;
  float b1 = (bias && has2) ? bias[n0 + 16 + lm] : 0.f;
#pragma unroll
  for (int r = 0; r < 8; ++r) {
    int m = m0 + r + hi * 8;
    float v0 = acc0[r] + b0;
    if (relu) v0 = v0 > 0.f ? v0 : 0.f;
    C[(size_t)m * N + n0 + lm] = v0;
    if (has2) {
      float v1 = acc1[r] + b1;
      if (relu) v1 = v1 > 0.f ? v1 : 0.f;
      C[(size_t)m * N + n0 + 16 + lm] = v1;
    }
  }
}

// ---------------------------------------------------------------------------
// Fused per-sample Jacobian (one workgroup / batch element, 8 waves):
//   J1^T = (h1>0) .* Wd1^T          (f16, LDS)
//   J2^T = (h2>0) .* (Wd2 @ J1)^T   (f16, LDS, WMMA)
//   W chunks = Wmu @ J2             (f16, LDS, WMMA; 784 = 384 + 400 rows)
//   S = W^T W (Gram, WMMA, in-register accumulation over chunks)
//   g = W^T (x - mu)
// ---------------------------------------------------------------------------
__global__ __launch_bounds__(256) void jacobian_kernel(
    const float* __restrict__ h1, const float* __restrict__ h2,
    const float* __restrict__ Wd1, const float* __restrict__ Wd2,
    const float* __restrict__ Wmu, const float* __restrict__ x,
    const float* __restrict__ mu, float* __restrict__ S,
    float* __restrict__ g) {
  __shared__ __attribute__((aligned(32))) _Float16 J1t[ZZ * HDIM];  // 32 KB
  __shared__ __attribute__((aligned(32))) _Float16 J2t[ZZ * HDIM];  // 32 KB
  _Float16* Wt = J1t;  // overlay: [ZZ][WSTRIDE] chunk (26 KB)

  int b = blockIdx.x;
  int tid = threadIdx.x;
  int lane = tid & 31, wv = tid >> 5;
  int lm = lane & 15, hi = lane >> 4;

  // ---- Stage 1: masked J1^T ----
  for (int j = tid; j < HDIM; j += 256) {
    float m = h1[(size_t)b * HDIM + j] > 0.f ? 1.f : 0.f;
    const v4f* wr = (const v4f*)(Wd1 + (size_t)j * ZZ);
#pragma unroll
    for (int q = 0; q < 8; ++q) {
      v4f w = wr[q];
#pragma unroll
      for (int e = 0; e < 4; ++e)
        J1t[(q * 4 + e) * HDIM + j] = (_Float16)(m * w[e]);
    }
  }
  __syncthreads();

  // ---- Stage 2: J2^T = mask2 .* (Wd2 @ J1), 16 rows x 32 z per wave ----
  for (int t = wv; t < 32; t += 8) {
    int i0 = t << 4;
    const float* arow = Wd2 + (size_t)(i0 + lm) * HDIM;
    const _Float16* b0 = J1t + lm * HDIM;
    const _Float16* b1 = J1t + (16 + lm) * HDIM;
    v8f acc0 = {}, acc1 = {};
    for (int k0 = 0; k0 < HDIM; k0 += 32) {
      __builtin_prefetch(arow + k0 + 64, 0, 0);
      v16h a = loadA_f32(arow, k0, hi);
      acc0 = wmma16(a, loadB_h(b0, k0, hi), acc0);
      acc1 = wmma16(a, loadB_h(b1, k0, hi), acc1);
    }
#pragma unroll
    for (int r = 0; r < 8; ++r) {
      int i = i0 + r + hi * 8;
      float m2 = h2[(size_t)b * HDIM + i] > 0.f ? 1.f : 0.f;
      J2t[lm * HDIM + i]        = (_Float16)(acc0[r] * m2);
      J2t[(16 + lm) * HDIM + i] = (_Float16)(acc1[r] * m2);
    }
  }
  __syncthreads();

  // ---- Stage 3: W chunks + Gram + g ----
  v8f accS = {};     // waves 0..3: one 16x16 tile of S
  float gacc = 0.f;  // wave 4, lane = z
  int ioff = 0;
#pragma unroll 1
  for (int p = 0; p < 2; ++p) {
    int clen = (p == 0) ? 384 : 400;
    int nti = clen >> 4;
    for (int t = wv; t < nti; t += 8) {
      int i0 = t << 4;
      const float* arow = Wmu + (size_t)(ioff + i0 + lm) * HDIM;
      const _Float16* b0 = J2t + lm * HDIM;
      const _Float16* b1 = J2t + (16 + lm) * HDIM;
      v8f acc0 = {}, acc1 = {};
      for (int k0 = 0; k0 < HDIM; k0 += 32) {
        __builtin_prefetch(arow + k0 + 64, 0, 0);
        v16h a = loadA_f32(arow, k0, hi);
        acc0 = wmma16(a, loadB_h(b0, k0, hi), acc0);
        acc1 = wmma16(a, loadB_h(b1, k0, hi), acc1);
      }
#pragma unroll
      for (int r = 0; r < 8; ++r) {
        Wt[lm * WSTRIDE + i0 + r + hi * 8]        = (_Float16)acc0[r];
        Wt[(16 + lm) * WSTRIDE + i0 + r + hi * 8] = (_Float16)acc1[r];
      }
    }
    if (clen < WSTRIDE) {  // zero pad columns [400, WSTRIDE)
      for (int idx = tid; idx < ZZ * (WSTRIDE - 400); idx += 256)
        Wt[(idx >> 4) * WSTRIDE + 400 + (idx & 15)] = (_Float16)0.f;
    }
    __syncthreads();

    int kend = (clen + 31) & ~31;  // 384 or 416 (tail zero-padded)
    if (wv < 4) {
      const _Float16* ar = Wt + (((wv >> 1) << 4) + lm) * WSTRIDE;
      const _Float16* br = Wt + (((wv & 1) << 4) + lm) * WSTRIDE;
      for (int k0 = 0; k0 < kend; k0 += 32)
        accS = wmma16(loadA_h(ar, k0, hi), loadB_h(br, k0, hi), accS);
    } else if (wv == 4) {
      const float* xb = x + (size_t)b * DDIM + ioff;
      const float* mb = mu + (size_t)b * DDIM + ioff;
      for (int i = 0; i < clen; i += 8) {
        v8h w8 = *(const v8h*)(Wt + lane * WSTRIDE + i);
        v4f xa = *(const v4f*)(xb + i), xc = *(const v4f*)(xb + i + 4);
        v4f ma = *(const v4f*)(mb + i), mc = *(const v4f*)(mb + i + 4);
#pragma unroll
        for (int e = 0; e < 4; ++e) {
          gacc += (float)w8[e] * (xa[e] - ma[e]);
          gacc += (float)w8[4 + e] * (xc[e] - mc[e]);
        }
      }
    }
    __syncthreads();
    ioff += clen;
  }

  if (wv < 4) {
    int z10 = (wv >> 1) << 4, z20 = (wv & 1) << 4;
#pragma unroll
    for (int r = 0; r < 8; ++r)
      S[(size_t)b * ZZ * ZZ + (z10 + r + hi * 8) * ZZ + z20 + lm] = accS[r];
  } else if (wv == 4) {
    g[(size_t)b * ZZ + lane] = gacc;
  }
}

// ---------------------------------------------------------------------------
// Per-sample 32x32 solve + mu mixing (one wave per sample, lane = row).
// cov = vinv*S + I ; rhs = vinv*(g + S@mu) ; Gaussian elimination (SPD).
// ---------------------------------------------------------------------------
__global__ __launch_bounds__(256) void solve_kernel(
    const float* __restrict__ S, const float* __restrict__ g,
    const float* __restrict__ dec_logvar, float* __restrict__ prev_mu,
    int iter) {
  __shared__ float aug[8][ZZ][ZZ + 2];
  int wv = threadIdx.x >> 5, lane = threadIdx.x & 31;
  int b = blockIdx.x * 8 + wv;
  float vinv = __expf(-dec_logvar[0]);
  float (*a)[ZZ + 2] = aug[wv];
  const float* Sb = S + (size_t)b * ZZ * ZZ;
  const float* pm = prev_mu + (size_t)b * ZZ;
  int i = lane;
  float rhsacc = 0.f;
  for (int j = 0; j < ZZ; ++j) {
    float sv = Sb[i * ZZ + j];
    a[i][j] = vinv * sv + (i == j ? 1.f : 0.f);
    rhsacc += vinv * sv * pm[j];
  }
  a[i][ZZ] = vinv * g[(size_t)b * ZZ + i] + rhsacc;
  for (int k = 0; k < ZZ; ++k) {
    float piv = a[k][k];
    float f = (i > k) ? a[i][k] / piv : 0.f;
    for (int j = k; j <= ZZ; ++j) {
      float akj = a[k][j];
      if (i > k) a[i][j] -= f * akj;
    }
  }
  float xv = 0.f;
  for (int k = ZZ - 1; k >= 0; --k) {
    float xk = a[k][ZZ] / a[k][k];
    if (i == k) xv = xk;
    if (i < k) a[i][ZZ] -= a[i][k] * xk;
  }
  float al = ALPHA_C / (1.f + (float)iter);
  prev_mu[(size_t)b * ZZ + i] = (1.f - al) * pm[i] + al * xv;
}

// ---------------------------------------------------------------------------
// Final: Gauss-Jordan inverse of cov, Cholesky of the inverse, z sample,
// per-sample (p_z - q_z). One wave per sample.
// ---------------------------------------------------------------------------
__global__ __launch_bounds__(64) void final_cov_kernel(
    const float* __restrict__ S, const float* __restrict__ dec_logvar,
    const float* __restrict__ prev_mu, const float* __restrict__ eps,
    float* __restrict__ z, float* __restrict__ t1) {
  __shared__ float GJ[2][ZZ][2 * ZZ + 1];
  __shared__ float CV[2][ZZ][ZZ + 1];
  __shared__ float DZ[2][ZZ];
  int wv = threadIdx.x >> 5, lane = threadIdx.x & 31;
  int b = blockIdx.x * 2 + wv;
  float vinv = __expf(-dec_logvar[0]);
  float (*A)[2 * ZZ + 1] = GJ[wv];
  float (*C)[ZZ + 1] = CV[wv];
  const float* Sb = S + (size_t)b * ZZ * ZZ;
  int i = lane;
  for (int j = 0; j < ZZ; ++j) {
    float cv = vinv * Sb[i * ZZ + j] + (i == j ? 1.f : 0.f);
    C[i][j] = cv;
    A[i][j] = cv;
    A[i][ZZ + j] = (i == j) ? 1.f : 0.f;
  }
  for (int k = 0; k < ZZ; ++k) {
    float inv = 1.f / A[k][k];
    if (i == k)
      for (int j = 0; j < 2 * ZZ; ++j) A[k][j] *= inv;
    float f = (i != k) ? A[i][k] : 0.f;
    for (int j = 0; j < 2 * ZZ; ++j) {
      float akj = A[k][j];
      if (i != k) A[i][j] -= f * akj;
    }
  }
  for (int k = 0; k < ZZ; ++k) {
    float lkk = sqrtf(fmaxf(A[k][ZZ + k], 1e-30f));
    if (i == k) A[k][ZZ + k] = lkk;
    if (i > k) A[i][ZZ + k] = A[i][ZZ + k] / lkk;
    if (i > k) {
      float lik = A[i][ZZ + k];
      for (int j = k + 1; j <= i; ++j) A[i][ZZ + j] -= lik * A[j][ZZ + k];
    }
  }
  const float* pm = prev_mu + (size_t)b * ZZ;
  const float* ep = eps + (size_t)b * ZZ;
  float dz_i = 0.f;
  for (int j = 0; j <= i; ++j) dz_i += A[i][ZZ + j] * ep[j];
  float zi = pm[i] + dz_i;
  z[(size_t)b * ZZ + i] = zi;
  DZ[wv][i] = dz_i;
  float r0 = logf(A[i][ZZ + i]);
  float qv = 0.f;
  for (int j = 0; j < ZZ; ++j) qv += C[i][j] * DZ[wv][j];
  float r1 = dz_i * qv;
  float r2 = 0.5f * (LOG2PI_C + zi * zi);
  for (int o = 16; o > 0; o >>= 1) {
    r0 += __shfl_xor(r0, o, 32);
    r1 += __shfl_xor(r1, o, 32);
    r2 += __shfl_xor(r2, o, 32);
  }
  if (i == 0) {
    float qz = -0.5f * ((float)ZZ * LOG2PI_C + 2.f * r0 + r1);
    t1[b] = r2 - qz;
  }
}

__global__ __launch_bounds__(256) void px_kernel(
    const float* __restrict__ x, const float* __restrict__ pxmu,
    const float* __restrict__ dec_logvar, const float* __restrict__ t1,
    float* __restrict__ tot) {
  int wv = threadIdx.x >> 5, lane = threadIdx.x & 31;
  int b = blockIdx.x * 8 + wv;
  float dlv = dec_logvar[0];
  float vinv = __expf(-dlv);
  const float* xb = x + (size_t)b * DDIM;
  const float* mb = pxmu + (size_t)b * DDIM;
  float s = 0.f;
  for (int d = lane; d < DDIM; d += 32) {
    float df = xb[d] - mb[d];
    s += LOG2PI_C + dlv + df * df * vinv;
  }
  for (int o = 16; o > 0; o >>= 1) s += __shfl_xor(s, o, 32);
  if (lane == 0) tot[b] = 0.5f * s + t1[b];
}

__global__ __launch_bounds__(256) void reduce_kernel(
    const float* __restrict__ tot, float* __restrict__ out, int B) {
  __shared__ float sm[256];
  float s = 0.f;
  for (int i = threadIdx.x; i < B; i += 256) s += tot[i];
  sm[threadIdx.x] = s;
  __syncthreads();
  for (int o = 128; o > 0; o >>= 1) {
    if ((int)threadIdx.x < o) sm[threadIdx.x] += sm[threadIdx.x + o];
    __syncthreads();
  }
  if (threadIdx.x == 0) out[0] = -(sm[0] / (float)B);
}

// ---------------------------------------------------------------------------
extern "C" void kernel_launch(void* const* d_in, const int* in_sizes, int n_in,
                              void* d_out, int out_size, void* d_ws,
                              size_t ws_size, hipStream_t stream) {
  (void)in_sizes; (void)n_in; (void)out_size; (void)ws_size;
  const float* x    = (const float*)d_in[0];
  const float* eps  = (const float*)d_in[1];
  const float* We1  = (const float*)d_in[2];
  const float* be1  = (const float*)d_in[3];
  const float* We2  = (const float*)d_in[4];
  const float* be2  = (const float*)d_in[5];
  const float* Wmue = (const float*)d_in[6];
  const float* bmue = (const float*)d_in[7];
  const float* Wd1  = (const float*)d_in[8];
  const float* bd1  = (const float*)d_in[9];
  const float* Wd2  = (const float*)d_in[10];
  const float* bd2  = (const float*)d_in[11];
  const float* Wmu  = (const float*)d_in[12];
  const float* bmu  = (const float*)d_in[13];
  const float* dlv  = (const float*)d_in[14];

  const int B = 2048, Dd = DDIM, Zz = ZZ, H = HDIM;
  float* ws = (float*)d_ws;
  size_t off = 0;
  float* he1   = ws + off; off += (size_t)B * H;
  float* he2   = ws + off; off += (size_t)B * H;
  float* pmu   = ws + off; off += (size_t)B * Zz;
  float* h1    = ws + off; off += (size_t)B * H;
  float* h2    = ws + off; off += (size_t)B * H;
  float* mu    = ws + off; off += (size_t)B * Dd;
  float* S     = ws + off; off += (size_t)B * Zz * Zz;
  float* g     = ws + off; off += (size_t)B * Zz;
  float* zb    = ws + off; off += (size_t)B * Zz;
  float* t1    = ws + off; off += (size_t)B;
  float* tot   = ws + off; off += (size_t)B;
  float* xpad  = ws + off; off += (size_t)B * KPAD;
  float* We1p  = ws + off; off += (size_t)H * KPAD;

  auto gemm = [&](const float* A, const float* W, const float* bias, float* C,
                  int M, int N, int K, int relu) {
    hipLaunchKernelGGL(gemm_wmma, dim3((M / 16) * ((N + 31) / 32)), dim3(32),
                       0, stream, A, W, bias, C, M, N, K, relu);
  };

  // pad K=784 operands to K=800 so all GEMM K-loops are unguarded
  hipLaunchKernelGGL(pad_copy, dim3(B), dim3(256), 0, stream, x, xpad, Dd, KPAD);
  hipLaunchKernelGGL(pad_copy, dim3(H), dim3(256), 0, stream, We1, We1p, Dd, KPAD);

  // encoder
  gemm(xpad, We1p, be1, he1, B, H, KPAD, 1);
  gemm(he1, We2, be2, he2, B, H, H, 1);
  gemm(he2, Wmue, bmue, pmu, B, Zz, H, 0);

  // Laplace mu updates
  for (int it = 0; it < 10; ++it) {
    gemm(pmu, Wd1, bd1, h1, B, H, Zz, 1);
    gemm(h1, Wd2, bd2, h2, B, H, H, 1);
    gemm(h2, Wmu, bmu, mu, B, Dd, H, 0);
    hipLaunchKernelGGL(jacobian_kernel, dim3(B), dim3(256), 0, stream,
                       h1, h2, Wd1, Wd2, Wmu, x, mu, S, g);
    hipLaunchKernelGGL(solve_kernel, dim3(B / 8), dim3(256), 0, stream,
                       S, g, dlv, pmu, it);
  }

  // final posterior cov + sample
  gemm(pmu, Wd1, bd1, h1, B, H, Zz, 1);
  gemm(h1, Wd2, bd2, h2, B, H, H, 1);
  gemm(h2, Wmu, bmu, mu, B, Dd, H, 0);
  hipLaunchKernelGGL(jacobian_kernel, dim3(B), dim3(256), 0, stream,
                     h1, h2, Wd1, Wd2, Wmu, x, mu, S, g);
  hipLaunchKernelGGL(final_cov_kernel, dim3(B / 2), dim3(64), 0, stream,
                     S, dlv, pmu, eps, zb, t1);

  // likelihood decoder pass on z
  gemm(zb, Wd1, bd1, h1, B, H, Zz, 1);
  gemm(h1, Wd2, bd2, h2, B, H, H, 1);
  gemm(h2, Wmu, bmu, mu, B, Dd, H, 0);

  hipLaunchKernelGGL(px_kernel, dim3(B / 8), dim3(256), 0, stream,
                     x, mu, dlv, t1, tot);
  hipLaunchKernelGGL(reduce_kernel, dim3(1), dim3(256), 0, stream,
                     tot, (float*)d_out, B);
}
